// MoTAttention_35656818491416
// MI455X (gfx1250) — matrix operand order repaired and checked by reading, not compiled, not run on hardware
//
#include <hip/hip_runtime.h>
#include <hip/hip_bf16.h>

// ---------------------------------------------------------------------------
// MoT attention for MI455X (gfx1250, wave32, WMMA), bf16 compute / fp32 accum.
//   K0a: weight swizzle: fp32 W -> bf16 packed in WMMA B-fragment layout
//        [m][colTile][kStep][lane][16] -> hot loops do 32B vector loads.
//   K0b: x -> per-modality masked bf16 copies (row-masked A operand), so the
//        projection inner loops carry no mask/convert VALU at all.
//   K1: modality-masked QKV projection + fused RoPE. Q/K row-major bf16,
//       V stored transposed (column-major) for the PV B-fragment.
//   K2: flash attention, wave-per-(b,h,16 q-rows), 32-token KV blocks,
//       online softmax, P transposed C->A layout via bf16 LDS (s_wait_dscnt).
//   K3: modality-masked output projection, fp32 result to d_out.
// ---------------------------------------------------------------------------

typedef __bf16 bf16x16 __attribute__((ext_vector_type(16)));
typedef float  f32x8   __attribute__((ext_vector_type(8)));
typedef unsigned short u16x8 __attribute__((ext_vector_type(8)));

static constexpr int BSZ = 2, SEQ = 2048, DIM = 2048;
static constexpr int H = 16, KVH = 4, HD = 128, MOD = 2;
static constexpr int NQ = H * HD;        // 2048
static constexpr int NK = KVH * HD;      // 512
static constexpr int NTOT = NQ + 2 * NK; // 3072
static constexpr int NTOK = BSZ * SEQ;   // 4096

__device__ __forceinline__ unsigned short f2bf_bits(float f) {
  __bf16 h = (__bf16)f;
  return __builtin_bit_cast(unsigned short, h);
}
__device__ __forceinline__ __bf16 bf_from_bits(unsigned short u) {
  return __builtin_bit_cast(__bf16, u);
}
// Load 8 contiguous bf16 (16B) into fragment elements [at, at+8).
__device__ __forceinline__ void load8(bf16x16& a, int at,
                                      const unsigned short* p) {
  u16x8 v = *reinterpret_cast<const u16x8*>(p);
#pragma unroll
  for (int j = 0; j < 8; ++j) a[at + j] = bf_from_bits(v[j]);
}
__device__ __forceinline__ bf16x16 load_frag32(const unsigned short* p) {
  return *reinterpret_cast<const bf16x16*>(p); // 32B aligned -> 2x b128
}

#define WMMA_BF16(A, B, C)                                                    \
  __builtin_amdgcn_wmma_f32_16x16x32_bf16(false, (A), false, (B), (short)0,   \
                                          (C), false, false)

// ===========================================================================
// Kernel 0a: weight swizzle fp32 -> bf16 B-fragment layout.
// out[(((m*(N/16)+ct)*(Kdim/32)+ks)*32+lane)*16+j] = W[m][ks*32+g*16+j][ct*16+hl]
// ===========================================================================
__global__ __launch_bounds__(256) void swizzle_w_kernel(
    const float* __restrict__ W, unsigned short* __restrict__ out, int N,
    int Kdim) {
  const int tid = blockIdx.x * blockDim.x + threadIdx.x;
  const int lane = tid & 31;
  int rest = tid >> 5;
  const int ks = rest % (Kdim / 32);
  rest /= (Kdim / 32);
  const int ct = rest % (N / 16);
  const int m = rest / (N / 16);
  const int g = lane >> 4, hl = lane & 15;
  const float* src =
      W + (size_t)m * Kdim * N + (size_t)(ks * 32 + g * 16) * N + ct * 16 + hl;
  u16x8 pack[2];
#pragma unroll
  for (int j = 0; j < 16; ++j)
    pack[j >> 3][j & 7] = f2bf_bits(src[(size_t)j * N]);
  u16x8* dst = reinterpret_cast<u16x8*>(out + (size_t)tid * 16);
  dst[0] = pack[0];
  dst[1] = pack[1];
}

// ===========================================================================
// Kernel 0b: x fp32 -> two modality-masked bf16 copies (8 elems/thread).
// ===========================================================================
__global__ __launch_bounds__(256) void mask_x_kernel(
    const float* __restrict__ x, const int* __restrict__ mids,
    unsigned short* __restrict__ xm0, unsigned short* __restrict__ xm1) {
  const int tid = blockIdx.x * blockDim.x + threadIdx.x;
  const int t = tid / (DIM / 8);
  const int c8 = tid % (DIM / 8);
  const int mod = mids[t];
  const float* p = x + (size_t)t * DIM + c8 * 8;
  u16x8 v0, v1;
#pragma unroll
  for (int j = 0; j < 8; ++j) {
    const unsigned short bb = f2bf_bits(p[j]);
    v0[j] = (mod == 0) ? bb : (unsigned short)0;
    v1[j] = (mod == 1) ? bb : (unsigned short)0;
  }
  const size_t o = (size_t)t * DIM + c8 * 8;
  *reinterpret_cast<u16x8*>(xm0 + o) = v0;
  *reinterpret_cast<u16x8*>(xm1 + o) = v1;
}

// ===========================================================================
// Kernel 1: masked QKV projection + RoPE.
// grid = (NTOK/16, NTOT/(16*8)), block = 256 (8 waves; wave -> one 16x16 tile)
// Inner loop: 2xb128 (A, pre-masked bf16) + 2xb128 (B, pre-swizzled) + WMMA.
// ===========================================================================
__global__ __launch_bounds__(256) void mot_qkv_rope_kernel(
    const unsigned short* __restrict__ xm0,
    const unsigned short* __restrict__ xm1, const float* __restrict__ fc,
    const unsigned short* __restrict__ qsw,
    const unsigned short* __restrict__ ksw,
    const unsigned short* __restrict__ vsw, unsigned short* __restrict__ qws,
    unsigned short* __restrict__ kws, unsigned short* __restrict__ vT) {
  const int lane = threadIdx.x & 31;
  const int wave = threadIdx.x >> 5;
  const int g = lane >> 4;    // lane half (K-split per ISA layout)
  const int hl = lane & 15;   // A-row / B-col within tile
  const int t0 = blockIdx.x * 16;
  const int c = (blockIdx.y * 8 + wave) * 16; // combined QKV column

  // Segment select (tile never crosses a segment: boundaries at 2048/2560).
  const unsigned short* wsw;
  int ldw, cseg, seg; // seg: 0=Q, 1=K, 2=V
  if (c < NQ) {
    wsw = qsw; ldw = NQ; cseg = c; seg = 0;
  } else if (c < NQ + NK) {
    wsw = ksw; ldw = NK; cseg = c - NQ; seg = 1;
  } else {
    wsw = vsw; ldw = NK; cseg = c - NQ - NK; seg = 2;
  }

  const int tA = t0 + hl; // this lane's A-matrix row (token)

  f32x8 acc = {0.f, 0.f, 0.f, 0.f, 0.f, 0.f, 0.f, 0.f};

#pragma unroll
  for (int m = 0; m < MOD; ++m) {
    const unsigned short* xrow =
        (m == 0 ? xm0 : xm1) + (size_t)tA * DIM + g * 16;
    // Pre-swizzled fragment stream for (m, colTile=cseg/16): 512 u16 per kstep.
    const unsigned short* bstream =
        wsw + (((size_t)m * (ldw / 16) + (cseg >> 4)) * (DIM / 32)) * 512 +
        lane * 16;
#pragma unroll 4
    for (int k = 0; k < DIM; k += 32) {
      bf16x16 a; // 16x32 bf16 A, contiguous-K (k = 16*g + s); rows pre-masked.
      load8(a, 0, xrow + k);
      load8(a, 8, xrow + k + 8);
      const bf16x16 b = load_frag32(bstream + (size_t)(k >> 5) * 512);
      acc = WMMA_BF16(a, b, acc);
    }
  }

  // Epilogue. C layout: row = i + 8*g, col = hl.
  if (seg != 2) {
    // Q/K: RoPE, row-major bf16 store (scatter across tokens).
    unsigned short* outp = (seg == 0) ? qws : kws;
#pragma unroll
    for (int i = 0; i < 8; ++i) {
      const int t = t0 + i + 8 * g;
      const int cc = cseg + hl;
      const float v = acc[i];
      const float partner = __shfl_xor(v, 1, 32); // paired column, same row
      const int hc = cc & (HD - 1);
      const int d = hc >> 1;
      const int s = t & (SEQ - 1);
      const float* f = fc + ((size_t)s * (HD / 2) + d) * 4;
      const float cs = f[0], sn = f[2]; // [[c,-s],[s,c]]
      const float outv =
          (hc & 1) ? (sn * partner + cs * v) : (cs * v - sn * partner);
      outp[(size_t)t * ldw + cc] = f2bf_bits(outv);
    }
  } else {
    // V: store transposed vT[(b*KVH+kvh)*HD + hc][s] -> packed 16B per lane.
    const int cv = cseg + hl;
    const int bb = t0 / SEQ;                 // whole tile in one batch
    const int s0 = (t0 & (SEQ - 1)) + 8 * g; // 8 consecutive tokens
    u16x8 pack;
#pragma unroll
    for (int i = 0; i < 8; ++i) pack[i] = f2bf_bits(acc[i]);
    unsigned short* dst =
        vT + ((size_t)(bb * KVH + (cv >> 7)) * HD + (cv & (HD - 1))) * SEQ + s0;
    *reinterpret_cast<u16x8*>(dst) = pack;
  }
}

// ===========================================================================
// Kernel 2: flash attention (causal, GQA). grid = (SEQ/16, H, BSZ), block = 32
// One wave owns a 16-row Q tile for one (b,h). KV consumed in 32-token blocks.
// ===========================================================================
__global__ __launch_bounds__(32) void mot_flash_attn_kernel(
    const unsigned short* __restrict__ qws,
    const unsigned short* __restrict__ kws,
    const unsigned short* __restrict__ vT, unsigned short* __restrict__ aws) {
  const int lane = threadIdx.x;
  const int g = lane >> 4, hl = lane & 15;
  const int qt = blockIdx.x, h = blockIdx.y, b = blockIdx.z;
  const int kvh = h >> 2; // REP = H/KVH = 4
  const float scale = 0.088388347648318447f; // 1/sqrt(128)

  const unsigned short* Qbase = qws + (size_t)(b * SEQ + qt * 16) * NQ + h * HD;
  const unsigned short* Kbase = kws + (size_t)(b * SEQ) * NK + kvh * HD;
  const unsigned short* VTbase = vT + (size_t)(b * KVH + kvh) * HD * SEQ;

  // Resident Q fragments: 4 chunks of K=32 (16x128 tile).
  bf16x16 qf[4];
#pragma unroll
  for (int kc = 0; kc < 4; ++kc) {
    const unsigned short* p = Qbase + (size_t)hl * NQ + kc * 32;
    load8(qf[kc], 0, p + g * 16);
    load8(qf[kc], 8, p + 8 + g * 16);
  }

  f32x8 o[8];
#pragma unroll
  for (int n = 0; n < 8; ++n)
#pragma unroll
    for (int i = 0; i < 8; ++i) o[n][i] = 0.f;
  float mrow[8], lrow[8];
#pragma unroll
  for (int i = 0; i < 8; ++i) { mrow[i] = -3.0e38f; lrow[i] = 0.f; }

  // bf16 P-transpose buffer; row stride 48 u16 keeps 16B alignment per group.
  __shared__ unsigned short sPb[16][48];

  const int nk = qt * 16 + 16; // causal KV extent
  for (int k0 = 0; k0 < nk; k0 += 32) {
    // ---- scores: two 16x16 tiles over this 32-token KV block ----
    f32x8 sc[2];
#pragma unroll
    for (int half = 0; half < 2; ++half) {
#pragma unroll
      for (int i = 0; i < 8; ++i) sc[half][i] = 0.f;
      const unsigned short* krow = Kbase + (size_t)(k0 + half * 16 + hl) * NK;
#pragma unroll
      for (int kc = 0; kc < 4; ++kc) {
        // B = K^T: B[k][n] = K[kt+n][k]; 32B contiguous per lane in k dim.
        const bf16x16 bf = load_frag32(krow + kc * 32 + g * 16);
        sc[half] = WMMA_BF16(qf[kc], bf, sc[half]);
      }
    }
    // ---- scale + causal mask ----
#pragma unroll
    for (int half = 0; half < 2; ++half) {
      const int kcol = k0 + half * 16 + hl;
#pragma unroll
      for (int i = 0; i < 8; ++i) {
        const int qrow = qt * 16 + i + 8 * g;
        sc[half][i] = (kcol <= qrow) ? sc[half][i] * scale : -3.0e38f;
      }
    }
    // ---- online softmax: rowwise max over 16 lanes of each half ----
    float mnew[8], alpha[8];
#pragma unroll
    for (int i = 0; i < 8; ++i) {
      float mx = fmaxf(sc[0][i], sc[1][i]);
      mx = fmaxf(mx, __shfl_xor(mx, 1, 32));
      mx = fmaxf(mx, __shfl_xor(mx, 2, 32));
      mx = fmaxf(mx, __shfl_xor(mx, 4, 32));
      mx = fmaxf(mx, __shfl_xor(mx, 8, 32));
      mnew[i] = fmaxf(mrow[i], mx);
      alpha[i] = __expf(mrow[i] - mnew[i]);
      mrow[i] = mnew[i];
    }
    // ---- probs, stash to LDS (bf16) for layout transpose ----
#pragma unroll
    for (int half = 0; half < 2; ++half)
#pragma unroll
      for (int i = 0; i < 8; ++i) {
        const float p = __expf(sc[half][i] - mnew[i]);
        sc[half][i] = p;
        sPb[i + 8 * g][half * 16 + hl] = f2bf_bits(p);
      }
#pragma unroll
    for (int i = 0; i < 8; ++i) {
      float rs = sc[0][i] + sc[1][i];
      rs += __shfl_xor(rs, 1, 32);
      rs += __shfl_xor(rs, 2, 32);
      rs += __shfl_xor(rs, 4, 32);
      rs += __shfl_xor(rs, 8, 32);
      lrow[i] = lrow[i] * alpha[i] + rs;
    }
#pragma unroll
    for (int n = 0; n < 8; ++n)
#pragma unroll
      for (int i = 0; i < 8; ++i) o[n][i] *= alpha[i];

    // LDS RAW across lanes within the wave: drain DScnt before A-frag reads.
    asm volatile("s_wait_dscnt 0x0" ::: "memory");

    // P as A fragment (16x32, contiguous-K layout): two 16B LDS vector loads.
    bf16x16 pa;
    load8(pa, 0, &sPb[hl][g * 16]);
    load8(pa, 8, &sPb[hl][8 + g * 16]);

    // ---- O += P @ V over 8 head-dim column chunks (vT: 32B per lane) ----
#pragma unroll
    for (int n = 0; n < 8; ++n) {
      const bf16x16 vb =
          load_frag32(VTbase + (size_t)(n * 16 + hl) * SEQ + k0 + g * 16);
      o[n] = WMMA_BF16(pa, vb, o[n]);
    }
  }

  // ---- finalize: O /= l, store bf16 attn[b, q, h*HD + col] ----
  unsigned short* outp = aws + (size_t)(b * SEQ + qt * 16) * NQ + h * HD;
#pragma unroll
  for (int n = 0; n < 8; ++n)
#pragma unroll
    for (int i = 0; i < 8; ++i) {
      const float val = o[n][i] / lrow[i];
      outp[(size_t)(i + 8 * g) * NQ + n * 16 + hl] = f2bf_bits(val);
    }
}

// ===========================================================================
// Kernel 3: masked output projection. grid=(NTOK/16, DIM/(16*8)), block=256
// ===========================================================================
__global__ __launch_bounds__(256) void mot_oproj_kernel(
    const unsigned short* __restrict__ aws, const int* __restrict__ mids,
    const unsigned short* __restrict__ osw, float* __restrict__ out) {
  const int lane = threadIdx.x & 31;
  const int wave = threadIdx.x >> 5;
  const int g = lane >> 4, hl = lane & 15;
  const int t0 = blockIdx.x * 16;
  const int c = (blockIdx.y * 8 + wave) * 16;

  const int tA = t0 + hl;
  const int myMod = mids[tA];
  const unsigned short* arow = aws + (size_t)tA * NQ + g * 16;
  const __bf16 bz = (__bf16)0.0f;

  f32x8 acc = {0.f, 0.f, 0.f, 0.f, 0.f, 0.f, 0.f, 0.f};
#pragma unroll
  for (int m = 0; m < MOD; ++m) {
    const bool amask = (myMod == m);
    const unsigned short* bstream =
        osw + (((size_t)m * (DIM / 16) + (c >> 4)) * (NQ / 32)) * 512 +
        lane * 16;
#pragma unroll 4
    for (int k = 0; k < NQ; k += 32) {
      bf16x16 a;
      load8(a, 0, arow + k);
      load8(a, 8, arow + k + 8);
      if (!amask) {
#pragma unroll
        for (int j = 0; j < 16; ++j) a[j] = bz;
      }
      const bf16x16 b = load_frag32(bstream + (size_t)(k >> 5) * 512);
      acc = WMMA_BF16(a, b, acc);
    }
  }
#pragma unroll
  for (int i = 0; i < 8; ++i)
    out[(size_t)(t0 + i + 8 * g) * DIM + c + hl] = acc[i];
}

// ===========================================================================
extern "C" void kernel_launch(void* const* d_in, const int* in_sizes, int n_in,
                              void* d_out, int out_size, void* d_ws,
                              size_t ws_size, hipStream_t stream) {
  const float* x = (const float*)d_in[0];
  const float* fc = (const float*)d_in[1];
  const int* mids = (const int*)d_in[2];
  const float* wq = (const float*)d_in[3];
  const float* wk = (const float*)d_in[4];
  const float* wv = (const float*)d_in[5];
  const float* wo = (const float*)d_in[6];
  float* out = (float*)d_out;

  // Workspace layout (bf16, bytes): qws 16M | kws 4M | vT 4M | aws 16M |
  //   qsw 16M | ksw 4M | vsw 4M | osw 16M | xm0 16M | xm1 16M  (~118MB)
  char* ws = (char*)d_ws;
  size_t off = 0;
  unsigned short* qws = (unsigned short*)(ws + off); off += (size_t)NTOK * NQ * 2;
  unsigned short* kws = (unsigned short*)(ws + off); off += (size_t)NTOK * NK * 2;
  unsigned short* vT  = (unsigned short*)(ws + off); off += (size_t)NTOK * NK * 2;
  unsigned short* aws = (unsigned short*)(ws + off); off += (size_t)NTOK * NQ * 2;
  unsigned short* qsw = (unsigned short*)(ws + off); off += (size_t)MOD * DIM * NQ * 2;
  unsigned short* ksw = (unsigned short*)(ws + off); off += (size_t)MOD * DIM * NK * 2;
  unsigned short* vsw = (unsigned short*)(ws + off); off += (size_t)MOD * DIM * NK * 2;
  unsigned short* osw = (unsigned short*)(ws + off); off += (size_t)MOD * NQ * DIM * 2;
  unsigned short* xm0 = (unsigned short*)(ws + off); off += (size_t)NTOK * DIM * 2;
  unsigned short* xm1 = (unsigned short*)(ws + off); off += (size_t)NTOK * DIM * 2;

  // K0a: weight swizzles (one thread per 32B output fragment slice).
  {
    int nThq = MOD * (NQ / 16) * (DIM / 32) * 32;  // wq: 524288
    int nThk = MOD * (NK / 16) * (DIM / 32) * 32;  // wk/wv: 131072
    int nTho = MOD * (DIM / 16) * (NQ / 32) * 32;  // wo: 524288
    swizzle_w_kernel<<<nThq / 256, 256, 0, stream>>>(wq, qsw, NQ, DIM);
    swizzle_w_kernel<<<nThk / 256, 256, 0, stream>>>(wk, ksw, NK, DIM);
    swizzle_w_kernel<<<nThk / 256, 256, 0, stream>>>(wv, vsw, NK, DIM);
    swizzle_w_kernel<<<nTho / 256, 256, 0, stream>>>(wo, osw, DIM, NQ);
  }
  // K0b: masked bf16 activations.
  mask_x_kernel<<<(NTOK * (DIM / 8)) / 256, 256, 0, stream>>>(x, mids, xm0,
                                                              xm1);

  dim3 g1(NTOK / 16, NTOT / 128);
  mot_qkv_rope_kernel<<<g1, 256, 0, stream>>>(xm0, xm1, fc, qsw, ksw, vsw, qws,
                                              kws, vT);
  dim3 g2(SEQ / 16, H, BSZ);
  mot_flash_attn_kernel<<<g2, 32, 0, stream>>>(qws, kws, vT, aws);
  dim3 g3(NTOK / 16, DIM / 128);
  mot_oproj_kernel<<<g3, 256, 0, stream>>>(aws, mids, osw, out);
}
